// BaichuanAttention_20830591385794
// MI455X (gfx1250) — compile-verified, matrix operand order checked by reading, not run on hardware
//
#include <hip/hip_runtime.h>
#include <hip/hip_bf16.h>

typedef __attribute__((ext_vector_type(16))) __bf16 v16bf;
typedef __attribute__((ext_vector_type(8)))  float  v8f;
typedef int v4i __attribute__((vector_size(16)));   // matches builtin param type

#define H_DIM   5120
#define NHEAD   40
#define HD      128
#define SEQ     2048
#define QKV_N   (3 * H_DIM)

// ---------------------------------------------------------------------------
// CDNA5 async global->LDS (ASYNCcnt path), with safe fallback.
// ---------------------------------------------------------------------------
#if defined(__has_builtin)
#if __has_builtin(__builtin_amdgcn_global_load_async_to_lds_b128)
#define HAVE_ASYNC_LDS 1
#endif
#endif
#ifndef HAVE_ASYNC_LDS
#define HAVE_ASYNC_LDS 0
#endif

__device__ __forceinline__ void copy16_to_lds(void* lds_dst, const void* gsrc) {
#if HAVE_ASYNC_LDS
    __builtin_amdgcn_global_load_async_to_lds_b128(
        (__attribute__((address_space(1))) v4i*)gsrc,
        (__attribute__((address_space(3))) v4i*)lds_dst, 0, 0);
#else
    *(float4*)lds_dst = *(const float4*)gsrc;
#endif
}

__device__ __forceinline__ void wait_async_lds() {
#if HAVE_ASYNC_LDS
#if __has_builtin(__builtin_amdgcn_s_wait_asynccnt)
    __builtin_amdgcn_s_wait_asynccnt(0);
#else
    asm volatile("s_wait_asynccnt 0x0" ::: "memory");
#endif
#endif
}

__device__ __forceinline__ v8f wmma_bf16(v16bf a, v16bf b, v8f c) {
    return __builtin_amdgcn_wmma_f32_16x16x32_bf16(
        false, a, false, b, (short)0, c, false, false);
}

// A/B 16x32 bf16 fragment K-index for element e (0..15), lane-half hi (0/1).
__device__ __forceinline__ int frag_k(int e, int hi) {
    return e + (e & 8) + hi * 8;
}

// ---------------------------------------------------------------------------
// GEMM tiling: BM=BN=128, BK=32, 256 threads = 8 waves, double-buffered LDS.
// ---------------------------------------------------------------------------
#define BM 128
#define BN 128
#define BK 32
#define LDS_STRIDE 40    // 32 + 8 bf16 pad -> conflict-free rows

// Kernel 1: proj = hidden x W_pack^T; scatter bf16 q/k/v [head][seq][hd].
// Grid: (M tiles, N tiles) so consecutive blocks share the same W tile (L2).
__global__ __launch_bounds__(256)
void gemm_qkv_kernel(const float* __restrict__ A,   // [SEQ][H_DIM]
                     const float* __restrict__ B,   // [QKV_N][H_DIM]
                     __bf16* __restrict__ qbuf,
                     __bf16* __restrict__ kbuf,
                     __bf16* __restrict__ vbuf) {
    __shared__ __bf16 As[2][BM * LDS_STRIDE];
    __shared__ __bf16 Bs[2][BN * LDS_STRIDE];

    const int tid  = threadIdx.x;
    const int lane = tid & 31;
    const int w    = tid >> 5;
    const int hi   = lane >> 4;
    const int ln   = lane & 15;
    const int blockM = blockIdx.x * BM;
    const int blockN = blockIdx.y * BN;
    const int waveM  = (w >> 1) * 32;
    const int waveN  = (w & 1) * 64;

    const int ldrow = tid >> 3;          // 0..31 (+32*i)
    const int ldcol = (tid & 7) * 4;

    float4 ra[4], rb[4];
    auto load_regs = [&](int k0) {
#pragma unroll
        for (int i = 0; i < 4; ++i) {
            int row = ldrow + 32 * i;
            ra[i] = *(const float4*)(A + (size_t)(blockM + row) * H_DIM + k0 + ldcol);
            rb[i] = *(const float4*)(B + (size_t)(blockN + row) * H_DIM + k0 + ldcol);
        }
    };
    auto store_lds = [&](int buf) {
#pragma unroll
        for (int i = 0; i < 4; ++i) {
            int row = ldrow + 32 * i;
            __bf16* da = &As[buf][row * LDS_STRIDE + ldcol];
            da[0] = (__bf16)ra[i].x; da[1] = (__bf16)ra[i].y;
            da[2] = (__bf16)ra[i].z; da[3] = (__bf16)ra[i].w;
            __bf16* db = &Bs[buf][row * LDS_STRIDE + ldcol];
            db[0] = (__bf16)rb[i].x; db[1] = (__bf16)rb[i].y;
            db[2] = (__bf16)rb[i].z; db[3] = (__bf16)rb[i].w;
        }
    };

    v8f acc[2][4];
#pragma unroll
    for (int i = 0; i < 2; ++i)
#pragma unroll
        for (int j = 0; j < 4; ++j) acc[i][j] = {};

    load_regs(0);
    store_lds(0);
    __syncthreads();

    const int nIter = H_DIM / BK;   // 160
    for (int it = 0; it < nIter; ++it) {
        const int cur = it & 1;
        const bool has_next = (it + 1) < nIter;
        if (has_next) load_regs((it + 1) * BK);   // overlap global with WMMA

        v16bf afrag[2], bfrag[4];
#pragma unroll
        for (int ti = 0; ti < 2; ++ti) {
            const __bf16* src = &As[cur][(waveM + ti * 16 + ln) * LDS_STRIDE];
#pragma unroll
            for (int e = 0; e < 16; ++e) afrag[ti][e] = src[frag_k(e, hi)];
        }
#pragma unroll
        for (int ni = 0; ni < 4; ++ni) {
            const __bf16* src = &Bs[cur][(waveN + ni * 16 + ln) * LDS_STRIDE];
#pragma unroll
            for (int e = 0; e < 16; ++e) bfrag[ni][e] = src[frag_k(e, hi)];
        }
#pragma unroll
        for (int ti = 0; ti < 2; ++ti)
#pragma unroll
            for (int ni = 0; ni < 4; ++ni)
                acc[ti][ni] = wmma_bf16(afrag[ti], bfrag[ni], acc[ti][ni]);

        if (has_next) store_lds(cur ^ 1);
        __syncthreads();
    }

    // Epilogue: sel/head/d are uniform per 16-col tile (16 | 128 | 5120 align).
#pragma unroll
    for (int ni = 0; ni < 4; ++ni) {
        const int nbase = blockN + waveN + ni * 16;
        const int sel   = nbase / H_DIM;
        const int rem   = nbase - sel * H_DIM;
        const int head  = rem >> 7;
        const int dbase = rem & 127;
        __bf16* dst = (sel == 0) ? qbuf : (sel == 1) ? kbuf : vbuf;
        __bf16* base = dst + (size_t)head * SEQ * HD + dbase + ln;
#pragma unroll
        for (int ti = 0; ti < 2; ++ti)
#pragma unroll
            for (int r = 0; r < 8; ++r) {
                int m = blockM + waveM + ti * 16 + r + hi * 8;
                base[(size_t)m * HD] = (__bf16)acc[ti][ni][r];
            }
    }
}

// ---------------------------------------------------------------------------
// Kernel 2: flash attention. Grid (SEQ/128, NHEAD), 8 waves x 16 query rows.
// K/V blocks (32x128 bf16) staged cooperatively in double-buffered LDS via
// async global->LDS; online softmax; P transposed C->A layout through LDS.
// ---------------------------------------------------------------------------
#define KV_STRIDE 136   // 128 + 8 bf16 pad

__global__ __launch_bounds__(256)
void flash_attn_kernel(const __bf16* __restrict__ qbuf,
                       const __bf16* __restrict__ kbuf,
                       const __bf16* __restrict__ vbuf,
                       const float* __restrict__ mask,   // [NHEAD][SEQ][SEQ]
                       __bf16* __restrict__ attnbuf) {   // [SEQ][H_DIM]
    __shared__ __bf16 Ks[2][32 * KV_STRIDE];
    __shared__ __bf16 Vs[2][32 * KV_STRIDE];
    __shared__ float  p_lds[8][16 * 32];

    const int tid  = threadIdx.x;
    const int lane = tid & 31;
    const int w    = tid >> 5;
    const int hi   = lane >> 4;
    const int ln   = lane & 15;
    const int h    = blockIdx.y;
    const int q0   = blockIdx.x * 128 + w * 16;

    const __bf16* Qh = qbuf + (size_t)h * SEQ * HD;
    const __bf16* Kh = kbuf + (size_t)h * SEQ * HD;
    const __bf16* Vh = vbuf + (size_t)h * SEQ * HD;
    float* pl = &p_lds[w][0];

    const int srow = tid >> 4;           // 0..15 (+16*i)
    const int scol = (tid & 15) * 8;     // 16B chunks along a 128-elem row
    auto stage_kv = [&](int kb, int buf) {
#pragma unroll
        for (int i = 0; i < 2; ++i) {
            int row = srow + 16 * i;
            copy16_to_lds(&Ks[buf][row * KV_STRIDE + scol],
                          Kh + (size_t)(kb + row) * HD + scol);
            copy16_to_lds(&Vs[buf][row * KV_STRIDE + scol],
                          Vh + (size_t)(kb + row) * HD + scol);
        }
    };

    // 16x128 Q block as 4 A-fragments (one-time global gather).
    v16bf qfrag[4];
    {
        const __bf16* qrow = Qh + (size_t)(q0 + ln) * HD;
#pragma unroll
        for (int c = 0; c < 4; ++c)
#pragma unroll
            for (int e = 0; e < 16; ++e)
                qfrag[c][e] = qrow[c * 32 + frag_k(e, hi)];
    }

    float mrow[8], lrow[8];
    v8f acc[8];
#pragma unroll
    for (int r = 0; r < 8; ++r) { mrow[r] = -3.0e38f; lrow[r] = 0.0f; }
#pragma unroll
    for (int t = 0; t < 8; ++t) acc[t] = {};

    const float sm_scale = 0.08838834764831845f;   // 1/sqrt(128)
    const int nIter = SEQ / 32;                    // 64

    stage_kv(0, 0);
    for (int it = 0; it < nIter; ++it) {
        const int kb  = it * 32;
        const int cur = it & 1;
        wait_async_lds();
        __syncthreads();                           // K/V[cur] ready for all waves
        if (it + 1 < nIter) stage_kv(kb + 32, cur ^ 1);   // overlap next DMA

        // S = Q * K^T -> two 16x16 tiles (keys kb..+15, kb+16..+31).
        v8f s0 = {}, s1 = {};
#pragma unroll
        for (int c = 0; c < 4; ++c) {
            v16bf b0, b1;
            const __bf16* kr0 = &Ks[cur][ln * KV_STRIDE + c * 32];
            const __bf16* kr1 = &Ks[cur][(16 + ln) * KV_STRIDE + c * 32];
#pragma unroll
            for (int e = 0; e < 16; ++e) {
                int kk = frag_k(e, hi);
                b0[e] = kr0[kk];
                b1[e] = kr1[kk];
            }
            s0 = wmma_bf16(qfrag[c], b0, s0);
            s1 = wmma_bf16(qfrag[c], b1, s1);
        }

        // Scale + mask + online softmax; rows live in VGPR r across 16 lanes.
#pragma unroll
        for (int r = 0; r < 8; ++r) {
            int m = q0 + r + hi * 8;
            const float* mp = mask + ((size_t)h * SEQ + m) * SEQ + kb + ln;
            if (kb + 32 < SEQ) __builtin_prefetch(mp + 32, 0, 0);
            float t0 = s0[r] * sm_scale + mp[0];
            float t1 = s1[r] * sm_scale + mp[16];
            float mx = fmaxf(t0, t1);
#pragma unroll
            for (int off = 1; off < 16; off <<= 1)
                mx = fmaxf(mx, __shfl_xor(mx, off, 32));
            float mnew = fmaxf(mrow[r], mx);
            float sc   = __expf(mrow[r] - mnew);
            mrow[r]    = mnew;
            float e0 = __expf(t0 - mnew);
            float e1 = __expf(t1 - mnew);
            float rs = e0 + e1;
#pragma unroll
            for (int off = 1; off < 16; off <<= 1)
                rs += __shfl_xor(rs, off, 32);
            lrow[r] = lrow[r] * sc + rs;
#pragma unroll
            for (int t = 0; t < 8; ++t) acc[t][r] *= sc;
            pl[(r + hi * 8) * 32 + ln]      = e0;   // P in C layout -> LDS
            pl[(r + hi * 8) * 32 + 16 + ln] = e1;
        }
        __syncthreads();                           // p_lds visible

        v16bf pfrag;
        {
            const float* prow = &pl[ln * 32];
#pragma unroll
            for (int e = 0; e < 16; ++e)
                pfrag[e] = (__bf16)prow[frag_k(e, hi)];
        }

        // acc += P * V (V from LDS; B-layout gather).
#pragma unroll
        for (int t = 0; t < 8; ++t) {
            v16bf vf;
#pragma unroll
            for (int e = 0; e < 16; ++e)
                vf[e] = Vs[cur][frag_k(e, hi) * KV_STRIDE + t * 16 + ln];
            acc[t] = wmma_bf16(pfrag, vf, acc[t]);
        }
        // No trailing barrier needed: next iter's top barrier orders
        // p_lds / Ks / Vs reuse (writes go to the other buffer).
    }

    // Normalize, store bf16 attn [seq][h*HD + d] for the o-proj GEMM.
#pragma unroll
    for (int r = 0; r < 8; ++r) {
        float inv = 1.0f / lrow[r];
        int m = q0 + r + hi * 8;
        __bf16* orow = attnbuf + (size_t)m * H_DIM + h * HD;
#pragma unroll
        for (int t = 0; t < 8; ++t)
            orow[t * 16 + ln] = (__bf16)(acc[t][r] * inv);
    }
}

// ---------------------------------------------------------------------------
// Kernel 3: out = attn_bf16 x o_proj^T (f32 out). A staged via async LDS copy.
// ---------------------------------------------------------------------------
__global__ __launch_bounds__(256)
void gemm_out_kernel(const __bf16* __restrict__ A,   // [SEQ][H_DIM] bf16
                     const float* __restrict__ B,    // [H_DIM][H_DIM] f32
                     float* __restrict__ out) {      // [SEQ][H_DIM] f32
    __shared__ __bf16 As[2][BM * LDS_STRIDE];
    __shared__ __bf16 Bs[2][BN * LDS_STRIDE];

    const int tid  = threadIdx.x;
    const int lane = tid & 31;
    const int w    = tid >> 5;
    const int hi   = lane >> 4;
    const int ln   = lane & 15;
    const int blockM = blockIdx.x * BM;
    const int blockN = blockIdx.y * BN;
    const int waveM  = (w >> 1) * 32;
    const int waveN  = (w & 1) * 64;

    // A: 512 chunks of 8 bf16 (16B); B: 1024 chunks of 4 f32 (16B, convert).
    const int arow = tid >> 2, acol = (tid & 3) * 8;
    const int brow = tid >> 3, bcol = (tid & 7) * 4;

    auto stage_a = [&](int k0, int buf) {
#pragma unroll
        for (int i = 0; i < 2; ++i) {
            int row = arow + 64 * i;
            copy16_to_lds(&As[buf][row * LDS_STRIDE + acol],
                          A + (size_t)(blockM + row) * H_DIM + k0 + acol);
        }
    };
    float4 rb[4];
    auto load_b = [&](int k0) {
#pragma unroll
        for (int i = 0; i < 4; ++i) {
            int row = brow + 32 * i;
            rb[i] = *(const float4*)(B + (size_t)(blockN + row) * H_DIM + k0 + bcol);
        }
    };
    auto store_b = [&](int buf) {
#pragma unroll
        for (int i = 0; i < 4; ++i) {
            int row = brow + 32 * i;
            __bf16* db = &Bs[buf][row * LDS_STRIDE + bcol];
            db[0] = (__bf16)rb[i].x; db[1] = (__bf16)rb[i].y;
            db[2] = (__bf16)rb[i].z; db[3] = (__bf16)rb[i].w;
        }
    };

    v8f acc[2][4];
#pragma unroll
    for (int i = 0; i < 2; ++i)
#pragma unroll
        for (int j = 0; j < 4; ++j) acc[i][j] = {};

    stage_a(0, 0);
    load_b(0);
    store_b(0);
    wait_async_lds();
    __syncthreads();

    const int nIter = H_DIM / BK;
    for (int it = 0; it < nIter; ++it) {
        const int cur = it & 1;
        const bool has_next = (it + 1) < nIter;
        if (has_next) {
            stage_a((it + 1) * BK, cur ^ 1);   // async DMA into other buffer
            load_b((it + 1) * BK);
        }

        v16bf afrag[2], bfrag[4];
#pragma unroll
        for (int ti = 0; ti < 2; ++ti) {
            const __bf16* src = &As[cur][(waveM + ti * 16 + ln) * LDS_STRIDE];
#pragma unroll
            for (int e = 0; e < 16; ++e) afrag[ti][e] = src[frag_k(e, hi)];
        }
#pragma unroll
        for (int ni = 0; ni < 4; ++ni) {
            const __bf16* src = &Bs[cur][(waveN + ni * 16 + ln) * LDS_STRIDE];
#pragma unroll
            for (int e = 0; e < 16; ++e) bfrag[ni][e] = src[frag_k(e, hi)];
        }
#pragma unroll
        for (int ti = 0; ti < 2; ++ti)
#pragma unroll
            for (int ni = 0; ni < 4; ++ni)
                acc[ti][ni] = wmma_bf16(afrag[ti], bfrag[ni], acc[ti][ni]);

        if (has_next) store_b(cur ^ 1);
        wait_async_lds();
        __syncthreads();
    }

#pragma unroll
    for (int ti = 0; ti < 2; ++ti)
#pragma unroll
        for (int ni = 0; ni < 4; ++ni)
#pragma unroll
            for (int r = 0; r < 8; ++r) {
                int m = blockM + waveM + ti * 16 + r + hi * 8;
                int n = blockN + waveN + ni * 16 + ln;
                out[(size_t)m * H_DIM + n] = acc[ti][ni][r];
            }
}

// ---------------------------------------------------------------------------
// Inputs: 0 input_pos (=arange, dead), 1 end (=SEQ, dead), 2 hidden_states,
// 3 attention_mask, 4 W_pack, 5 o_proj_w, 6/7 caches (zeros, dead).
// ---------------------------------------------------------------------------
extern "C" void kernel_launch(void* const* d_in, const int* in_sizes, int n_in,
                              void* d_out, int out_size, void* d_ws, size_t ws_size,
                              hipStream_t stream) {
    const float* hidden = (const float*)d_in[2];
    const float* mask   = (const float*)d_in[3];
    const float* wpack  = (const float*)d_in[4];
    const float* oproj  = (const float*)d_in[5];
    float* out = (float*)d_out;

    const size_t headbuf_bytes = (size_t)NHEAD * SEQ * HD * sizeof(__bf16);  // 20 MB
    char* ws = (char*)d_ws;
    __bf16* qbuf = (__bf16*)(ws);
    __bf16* kbuf = (__bf16*)(ws + headbuf_bytes);
    __bf16* vbuf = (__bf16*)(ws + 2 * headbuf_bytes);
    __bf16* attn = (__bf16*)(ws + 3 * headbuf_bytes);

    dim3 g1(SEQ / BM, QKV_N / BN);   // (16, 120): M fastest -> W tile L2 reuse
    gemm_qkv_kernel<<<g1, 256, 0, stream>>>(hidden, wpack, qbuf, kbuf, vbuf);

    dim3 g2(SEQ / 128, NHEAD);       // (16, 40): q-blocks of a head adjacent
    flash_attn_kernel<<<g2, 256, 0, stream>>>(qbuf, kbuf, vbuf, mask, attn);

    dim3 g3(SEQ / BM, H_DIM / BN);   // (16, 40)
    gemm_out_kernel<<<g3, 256, 0, stream>>>(attn, oproj, out);
}
